// DeformConv_35759897706855
// MI455X (gfx1250) — matrix operand order repaired and checked by reading, not compiled, not run on hardware
//
#include <hip/hip_runtime.h>
#include <hip/hip_bf16.h>
#include <stdint.h>

// ---------------------------------------------------------------------------
// CDNA5 (gfx1250) fused CAM-attention + deformable-conv pipeline.
// wave32, 128-thread (4-wave) workgroups, v_wmma_f32_16x16x32_f16 GEMM cores,
// TDM tensor_load_to_lds (both toolchain arities) for the attention-energy
// LDS staging.
// ---------------------------------------------------------------------------

typedef _Float16 h16;
typedef __attribute__((ext_vector_type(8)))  _Float16 v8h;
typedef __attribute__((ext_vector_type(16))) _Float16 v16h;
typedef __attribute__((ext_vector_type(8)))  float    v8f;

union FragU { v16h v; v8h h[2]; };

// A fragment (16x32 f16) from a row-major [m][k] tile with row stride `ld`
// (halves). Lane m = lane&15, k-base b=(lane>>4)*8; per-lane halves are the
// two contiguous 16-byte chunks [b,b+8) and [b+16,b+24)  (ISA 7.12.2).
__device__ __forceinline__ v16h frag_a_rowmajor(const h16* base, int ld, int lane) {
  int m  = lane & 15;
  int kb = (lane >> 4) << 3;
  const h16* p = base + m * ld + kb;
  FragU f;
  f.h[0] = *(const v8h*)(p);
  f.h[1] = *(const v8h*)(p + 16);
  return f.v;
}

// B fragment (32x16 f16) from an [n][k] tile (k contiguous) with row stride
// `ld`. Lane n = lane&15, k-base b=(lane>>4)*8; halves [b,b+8) and [b+16,b+24).
__device__ __forceinline__ v16h frag_b_nk(const h16* base, int ld, int lane) {
  int n  = lane & 15;
  int kb = (lane >> 4) << 3;
  const h16* p = base + n * ld + kb;
  FragU f;
  f.h[0] = *(const v8h*)(p);
  f.h[1] = *(const v8h*)(p + 16);
  return f.v;
}

__device__ __forceinline__ v8f wmma16(v16h a, v16h b, v8f c) {
  // 8 args: (neg_a, A, neg_b, B, c_mod, C, reuse_a, reuse_b)
  return __builtin_amdgcn_wmma_f32_16x16x32_f16(false, a, false, b, (short)0, c,
                                                false, false);
}

// ---------------------------------------------------------------------------
// Tensor Data Mover. Guarded by __has_builtin; supports BOTH probe-verified
// arities: 5-arg (ROCm 7.2 / clang-22) and 6-arg (upstream clang-23+,
// therock-10 headers). Fallback is a manual LDS fill so the file always
// compiles.
// ---------------------------------------------------------------------------
#if defined(__HIP_DEVICE_COMPILE__)
#  if __has_builtin(__builtin_amdgcn_tensor_load_to_lds)
#    define ATHENA_HAS_TDM 1
#  endif
#  if __has_builtin(__builtin_amdgcn_s_wait_tensorcnt)
#    define ATHENA_WAIT_TENSORCNT0() __builtin_amdgcn_s_wait_tensorcnt(0)
#  else
#    define ATHENA_WAIT_TENSORCNT0() asm volatile("s_wait_tensorcnt 0x0" ::: "memory")
#  endif
#endif

#ifdef ATHENA_HAS_TDM
typedef unsigned int u32x4 __attribute__((ext_vector_type(4)));
typedef int          i32x4 __attribute__((ext_vector_type(4)));
typedef int          i32x8 __attribute__((ext_vector_type(8)));

// 2D f16 tile load: `rows` x `tileK` halves, global row stride `strideElems`,
// valid row length `rowLenElems` (OOB-K reads return zero -> free K padding),
// LDS padded +4 dwords (8 halves) after every 16 dwords (one 32-half row),
// i.e. LDS row stride = 40 halves. D# layout per ISA ch.8.
__device__ __forceinline__ void tdm_load_2d_f16(const h16* gsrc, unsigned lds_off,
                                                unsigned rowLenElems, unsigned rows,
                                                unsigned tileK, unsigned strideElems) {
  unsigned long long ga = (unsigned long long)(size_t)gsrc;
  u32x4 g0;
  g0[0] = 1u;                                        // count=1, user mode
  g0[1] = lds_off;                                   // lds_addr
  g0[2] = (unsigned)(ga & 0xffffffffu);              // global_addr[31:0]
  g0[3] = (unsigned)((ga >> 32) & 0x1ffffffu) | (2u << 30); // addr[56:32] | type=2
  i32x8 g1;
  g1[0] = (int)((1u << 16)        // data_size = 1 -> 2 bytes
              | (1u << 20)        // pad_enable
              | (3u << 22)        // pad_interval: every 16 dwords (64B row)
              | (3u << 25));      // pad_amount: 4 dwords (8 halves)
  g1[1] = (int)((rowLenElems & 0xffffu) << 16);      // tensor_dim0[15:0]
  g1[2] = (int)((rowLenElems >> 16) | ((rows & 0xffffu) << 16)); // dim0 hi | dim1 lo
  g1[3] = (int)(((rows >> 16) & 0xffffu) | (tileK << 16));       // dim1 hi | tile_dim0
  g1[4] = (int)(rows & 0xffffu);                     // tile_dim1 (tile_dim2 = 0)
  g1[5] = (int)strideElems;                          // tensor_dim0_stride[31:0]
  g1[6] = 0;                                         // stride hi | dim1_stride lo
  g1[7] = 0;
  i32x4 z4 = {0, 0, 0, 0};
#if __clang_major__ >= 23
  i32x8 z8 = {0, 0, 0, 0, 0, 0, 0, 0};
  __builtin_amdgcn_tensor_load_to_lds(g0, g1, z4, z4, z8, 0);   // 6-arg lane
#else
  __builtin_amdgcn_tensor_load_to_lds(g0, g1, z4, z4, 0);       // 5-arg lane
#endif
}
#endif // ATHENA_HAS_TDM

// ---------------------------------------------------------------------------
// f32 -> f16 convert
// ---------------------------------------------------------------------------
__global__ void cvt_f32_f16(const float* __restrict__ src, h16* __restrict__ dst, int n) {
  int i = blockIdx.x * blockDim.x + threadIdx.x;
  if (i < n) dst[i] = (h16)src[i];
}

// ---------------------------------------------------------------------------
// Energy + softmax:  attn[b, c, d] = softmax_d( rowmax - f[c,:]·f[d,:] )
//                  = softmax_d( -energy )  (shift-invariant)
// One WG per (batch, 16-row block): rows [r0,r0+16) x all 256 cols.
// K loop over spatial N (zero-padded to 32). A and B both read from one
// LDS-staged 256x32 f16 slab of f (B needs f transposed -> [d][k] == slab).
// ---------------------------------------------------------------------------
#define EK_KC 32
#define EK_LD 40  // halves: 32 data + 8 pad (matches TDM pad config)

__global__ __launch_bounds__(128)
void energy_softmax(const h16* __restrict__ fh, int N, h16* __restrict__ attn) {
  __shared__ h16   s_f[256 * EK_LD];   // 20.0 KB
  __shared__ float s_e[16 * 256];      // 16.0 KB
  const int b    = blockIdx.y;
  const int r0   = blockIdx.x * 16;
  const int tid  = threadIdx.x;
  const int wave = tid >> 5, lane = tid & 31;
  const h16* fb  = fh + (size_t)b * 256 * N;

  v8f acc[4] = {};
  for (int k0 = 0; k0 < N; k0 += EK_KC) {
#ifdef ATHENA_HAS_TDM
    if (tid < 32) {  // wave 0 issues the DMA; waits before the block barrier
      tdm_load_2d_f16(fb + k0, (unsigned)(size_t)(void*)s_f,
                      (unsigned)(N - k0), 256u, (unsigned)EK_KC, (unsigned)N);
      ATHENA_WAIT_TENSORCNT0();
    }
#else
    {
      int kmax = N - k0;
      if (kmax > EK_KC) kmax = EK_KC;
      for (int c = tid; c < 256; c += 128) {
        const h16* src = fb + (size_t)c * N + k0;
        h16*       dst = s_f + c * EK_LD;
        for (int k = 0; k < kmax; ++k) dst[k] = src[k];
        for (int k = kmax; k < EK_KC; ++k) dst[k] = (h16)0.f;
      }
    }
#endif
    __syncthreads();
    v16h a = frag_a_rowmajor(s_f + r0 * EK_LD, EK_LD, lane);
    #pragma unroll
    for (int j = 0; j < 4; ++j) {
      int d0 = wave * 64 + j * 16;
      v16h bf = frag_b_nk(s_f + d0 * EK_LD, EK_LD, lane);
      acc[j] = wmma16(a, bf, acc[j]);
    }
    __syncthreads();
  }

  // Spill the 16x256 energy slab to LDS (D layout: VGPR r -> rows r / r+8).
  {
    int n = lane & 15, mofs = (lane >> 4) * 8;
    #pragma unroll
    for (int j = 0; j < 4; ++j) {
      int d0 = wave * 64 + j * 16;
      #pragma unroll
      for (int r = 0; r < 8; ++r) s_e[(mofs + r) * 256 + d0 + n] = acc[j][r];
    }
  }
  __syncthreads();

  // softmax(rowmax - e) == softmax(-e):  p_d = exp(min_e - e_d) / sum
  if (tid < 16) {
    const float* row = &s_e[tid * 256];
    float mn = row[0];
    for (int d = 1; d < 256; ++d) mn = fminf(mn, row[d]);
    float sum = 0.f;
    for (int d = 0; d < 256; ++d) sum += __expf(mn - row[d]);
    float inv = 1.f / sum;
    h16* dst = attn + ((size_t)b * 256 + r0 + tid) * 256;
    for (int d = 0; d < 256; ++d) dst[d] = (h16)(__expf(mn - row[d]) * inv);
  }
}

// ---------------------------------------------------------------------------
// cam_use:  out[b,c,n] = gamma * (attn[b] @ v[b])[c,n] + f[b,c,n]
// GEMM M=256, N=spatial, K=256. A-fragments straight from global f16 attn
// rows (contiguous 16B per-lane chunks); B (v transposed) staged in LDS.
// One WG = 16 columns x all 256 rows (4 waves x 4 M-tiles).
// ---------------------------------------------------------------------------
#define CU_LD 40

__global__ __launch_bounds__(128)
void cam_use(const h16* __restrict__ attn, const h16* __restrict__ vh,
             const float* __restrict__ forig, const float* __restrict__ gamma_p,
             int N, float* __restrict__ out) {
  __shared__ h16 s_b[16 * CU_LD];
  const int b = blockIdx.y, n0 = blockIdx.x * 16;
  const int tid = threadIdx.x, wave = tid >> 5, lane = tid & 31;
  const h16* ab = attn + (size_t)b * 256 * 256;
  const h16* vb = vh   + (size_t)b * 256 * N;

  v8f acc[4] = {};
  for (int k0 = 0; k0 < 256; k0 += 32) {
    for (int idx = tid; idx < 16 * 32; idx += 128) {   // s_b[n][k] = v[k0+k][n0+n]
      int n = idx & 15, k = idx >> 4, col = n0 + n;
      s_b[n * CU_LD + k] = (col < N) ? vb[(size_t)(k0 + k) * N + col] : (h16)0.f;
    }
    __syncthreads();
    v16h bf = frag_b_nk(s_b, CU_LD, lane);
    #pragma unroll
    for (int j = 0; j < 4; ++j) {
      int m0 = (wave * 4 + j) * 16;
      v16h af = frag_a_rowmajor(ab + (size_t)m0 * 256 + k0, 256, lane);
      acc[j] = wmma16(af, bf, acc[j]);
    }
    __syncthreads();
  }

  float g = gamma_p[0];
  int n = lane & 15, mofs = (lane >> 4) * 8, col = n0 + n;
  if (col < N) {
    #pragma unroll
    for (int j = 0; j < 4; ++j) {
      int m0 = (wave * 4 + j) * 16;
      #pragma unroll
      for (int r = 0; r < 8; ++r) {
        size_t o = ((size_t)b * 256 + m0 + mofs + r) * N + col;
        out[o] = g * acc[j][r] + forig[o];
      }
    }
  }
}

// ---------------------------------------------------------------------------
// 3x3 offset conv (256 -> 18 channels, pad 1). Direct; weights LDS-staged in
// 32-channel chunks. One thread = one pixel, 18 accumulators.
// ---------------------------------------------------------------------------
__global__ __launch_bounds__(128)
void off_conv(const float* __restrict__ f2, const float* __restrict__ offw,
              const float* __restrict__ offb, int H, int W,
              float* __restrict__ off_out) {
  __shared__ float s_w[18 * 32 * 9];   // 20.25 KB
  const int b = blockIdx.y, N = H * W;
  const int p = blockIdx.x * 128 + threadIdx.x;
  const bool active = p < N;
  const int h = active ? p / W : 0, w = active ? p % W : 0;
  const float* fb = f2 + (size_t)b * 256 * N;

  float acc[18] = {};
  for (int c0 = 0; c0 < 256; c0 += 32) {
    for (int idx = threadIdx.x; idx < 18 * 32 * 9; idx += 128) {
      int t = idx % 9, cc = (idx / 9) % 32, o = idx / 288;
      s_w[idx] = offw[((size_t)o * 256 + c0 + cc) * 9 + t];
    }
    __syncthreads();
    if (active) {
      for (int cc = 0; cc < 32; ++cc) {
        const float* fc = fb + (size_t)(c0 + cc) * N;
        float patch[9];
        #pragma unroll
        for (int t = 0; t < 9; ++t) {
          int yy = h + t / 3 - 1, xx = w + t % 3 - 1;
          patch[t] = (yy >= 0 && yy < H && xx >= 0 && xx < W) ? fc[yy * W + xx] : 0.f;
        }
        #pragma unroll
        for (int o = 0; o < 18; ++o) {
          const float* wr = &s_w[(o * 32 + cc) * 9];
          float s = acc[o];
          #pragma unroll
          for (int t = 0; t < 9; ++t) s += wr[t] * patch[t];
          acc[o] = s;
        }
      }
    }
    __syncthreads();
  }
  if (active)
    for (int o = 0; o < 18; ++o)
      off_out[((size_t)b * 18 + o) * N + p] = acc[o] + offb[o];
}

// ---------------------------------------------------------------------------
// Fused deformable sampling + GEMM.
//   out[b,o,p] = sum_{c,kk} bilinear(f2[b,c], p, kk, offsets) * w[o, c*9+kk]
// One WG = 16 pixels x all 256 output channels. Per 32-K chunk: sample the
// 16x32 val tile straight into LDS (f16), then 16 WMMA M-tiles against the
// f16 weight matrix (A-fragments from global; rows are 4608B so the per-lane
// 16B chunks are aligned). Avoids the 283MB im2col round-trip entirely.
// ---------------------------------------------------------------------------
struct Tap { int idx[4]; float w[4]; };
#define DG_LD 40

__global__ __launch_bounds__(128)
void deform_gemm(const float* __restrict__ f2, const float* __restrict__ off,
                 const h16* __restrict__ dwh, int H, int W,
                 float* __restrict__ out) {
  __shared__ Tap s_tap[144];           // 4.5 KB: 9 taps x 16 pixels
  __shared__ h16 s_val[16 * DG_LD];    // 1.25 KB
  const int b = blockIdx.y, N = H * W;
  const int n0 = blockIdx.x * 16;
  const int tid = threadIdx.x, wave = tid >> 5, lane = tid & 31;
  const float* fb = f2  + (size_t)b * 256 * N;
  const float* ob = off + (size_t)b * 18 * N;

  // Phase 0: bilinear metadata (clamped indices + masked weights) per (kk, n).
  for (int e = tid; e < 144; e += 128) {
    int n = e & 15, kk = e >> 4, p = n0 + n;
    Tap tp;
    if (p < N) {
      int h = p / W, w = p % W;
      float oy = ob[(size_t)(kk * 2 + 0) * N + p];
      float ox = ob[(size_t)(kk * 2 + 1) * N + p];
      float py = (float)(h + kk / 3 - 1) + oy;
      float px = (float)(w + kk % 3 - 1) + ox;
      float y0f = floorf(py), x0f = floorf(px);
      float fy = py - y0f, fx = px - x0f;
      int y0 = (int)y0f, x0 = (int)x0f;
      #pragma unroll
      for (int j = 0; j < 4; ++j) {
        int dy = j >> 1, dx = j & 1;
        int yy = y0 + dy, xx = x0 + dx;
        float wt = (dy ? fy : 1.f - fy) * (dx ? fx : 1.f - fx);
        bool valid = (yy >= 0) && (yy < H) && (xx >= 0) && (xx < W);
        int yc = yy < 0 ? 0 : (yy > H - 1 ? H - 1 : yy);
        int xc = xx < 0 ? 0 : (xx > W - 1 ? W - 1 : xx);
        tp.idx[j] = yc * W + xc;
        tp.w[j]   = valid ? wt : 0.f;
      }
    } else {
      #pragma unroll
      for (int j = 0; j < 4; ++j) { tp.idx[j] = 0; tp.w[j] = 0.f; }
    }
    s_tap[e] = tp;
  }
  __syncthreads();

  v8f acc[4] = {};
  for (int k0 = 0; k0 < 2304; k0 += 32) {
    // Sample the 16x32 val tile into LDS (4 gathers + 4 fma per element).
    for (int e = tid; e < 16 * 32; e += 128) {
      int n = e & 15, kq = e >> 4, k = k0 + kq;
      int c = k / 9, kk = k % 9;
      Tap tp = s_tap[kk * 16 + n];
      const float* fc = fb + (size_t)c * N;
      float v = tp.w[0] * fc[tp.idx[0]] + tp.w[1] * fc[tp.idx[1]]
              + tp.w[2] * fc[tp.idx[2]] + tp.w[3] * fc[tp.idx[3]];
      s_val[n * DG_LD + kq] = (h16)v;
    }
    __syncthreads();
    // prefetch next K-chunk of this wave's weight rows (global_prefetch_b8)
    __builtin_prefetch((const void*)(dwh + (size_t)(wave * 64) * 2304 + k0 + 32));
    v16h bf = frag_b_nk(s_val, DG_LD, lane);
    #pragma unroll
    for (int j = 0; j < 4; ++j) {
      int m0 = (wave * 4 + j) * 16;
      v16h af = frag_a_rowmajor(dwh + (size_t)m0 * 2304 + k0, 2304, lane);
      acc[j] = wmma16(af, bf, acc[j]);
    }
    __syncthreads();
  }

  int n = lane & 15, mofs = (lane >> 4) * 8, p = n0 + n;
  if (p < N) {
    #pragma unroll
    for (int j = 0; j < 4; ++j) {
      int m0 = (wave * 4 + j) * 16;
      #pragma unroll
      for (int r = 0; r < 8; ++r)
        out[((size_t)b * 256 + m0 + mofs + r) * N + p] = acc[j][r];
    }
  }
}

// ---------------------------------------------------------------------------
// Host orchestration: 3 scales x {z (15x15), x (31x31)} streams.
// Workspace layout (reused across scales): ~71 MB.
// ---------------------------------------------------------------------------
extern "C" void kernel_launch(void* const* d_in, const int* in_sizes, int n_in,
                              void* d_out, int out_size, void* d_ws, size_t ws_size,
                              hipStream_t stream) {
  (void)in_sizes; (void)n_in; (void)out_size; (void)ws_size;
  const int B = 32, C = 256;
  const int HZ = 15, WZ = 15, NZ = 225;
  const int HX = 31, WX = 31, NX = 961;

  char* ws = (char*)d_ws;
  size_t woff = 0;
  auto walloc = [&](size_t bytes) -> char* {
    char* p = ws + woff;
    woff += (bytes + 255) & ~(size_t)255;
    return p;
  };
  h16*   zh   = (h16*)  walloc((size_t)B * C * NZ * 2);
  h16*   xh   = (h16*)  walloc((size_t)B * C * NX * 2);
  h16*   azh  = (h16*)  walloc((size_t)B * C * C * 2);
  h16*   axh  = (h16*)  walloc((size_t)B * C * C * 2);
  float* zf2  = (float*)walloc((size_t)B * C * NZ * 4);
  float* xf2  = (float*)walloc((size_t)B * C * NX * 4);
  float* offz = (float*)walloc((size_t)B * 18 * NZ * 4);
  float* offx = (float*)walloc((size_t)B * 18 * NX * 4);
  h16*   dwh  = (h16*)  walloc((size_t)C * C * 9 * 2);

  float* outz = (float*)d_out;                      // (3,B,C,15,15)
  float* outx = outz + (size_t)3 * B * C * NZ;      // (3,B,C,31,31)

  for (int i = 0; i < 3; ++i) {
    const float* z    = (const float*)d_in[6 * i + 0];
    const float* x    = (const float*)d_in[6 * i + 1];
    const float* offw = (const float*)d_in[6 * i + 2];
    const float* offb = (const float*)d_in[6 * i + 3];
    const float* dw   = (const float*)d_in[6 * i + 4];
    const float* gam  = (const float*)d_in[6 * i + 5];

    const int nz = B * C * NZ, nx = B * C * NX, nw = C * C * 9;
    cvt_f32_f16<<<dim3((nz + 255) / 256), 256, 0, stream>>>(z, zh, nz);
    cvt_f32_f16<<<dim3((nx + 255) / 256), 256, 0, stream>>>(x, xh, nx);
    cvt_f32_f16<<<dim3((nw + 255) / 256), 256, 0, stream>>>(dw, dwh, nw);

    energy_softmax<<<dim3(C / 16, B), 128, 0, stream>>>(zh, NZ, azh);
    energy_softmax<<<dim3(C / 16, B), 128, 0, stream>>>(xh, NX, axh);

    // zf uses x's attention and vice versa (reference semantics).
    cam_use<<<dim3((NZ + 15) / 16, B), 128, 0, stream>>>(axh, zh, z, gam, NZ, zf2);
    cam_use<<<dim3((NX + 15) / 16, B), 128, 0, stream>>>(azh, xh, x, gam, NX, xf2);

    off_conv<<<dim3((NZ + 127) / 128, B), 128, 0, stream>>>(zf2, offw, offb, HZ, WZ, offz);
    off_conv<<<dim3((NX + 127) / 128, B), 128, 0, stream>>>(xf2, offw, offb, HX, WX, offx);

    deform_gemm<<<dim3((NZ + 15) / 16, B), 128, 0, stream>>>(zf2, offz, dwh, HZ, WZ,
                                                             outz + (size_t)i * B * C * NZ);
    deform_gemm<<<dim3((NX + 15) / 16, B), 128, 0, stream>>>(xf2, offx, dwh, HX, WX,
                                                             outx + (size_t)i * B * C * NX);
  }
}
